// WaveNet_38465727103528
// MI455X (gfx1250) — matrix operand order repaired and checked by reading, not compiled, not run on hardware
//
#include <hip/hip_runtime.h>
#include <math.h>

#define B_     2
#define T_     19200
#define H_     64
#define SKIP_  256
#define NL_    24
#define LF_    80
#define L_     64
#define MU1_   256
#define TTILE  16
#define NTBLK  (T_ / TTILE)        // 1200 time tiles per batch

typedef __attribute__((ext_vector_type(16))) _Float16 v16h;
typedef __attribute__((ext_vector_type(8)))  _Float16 v8h;
typedef __attribute__((ext_vector_type(4)))  _Float16 v4h;
typedef __attribute__((ext_vector_type(8)))  float    v8f;

// LDS strides (elements) -- chosen so every fragment address is 32B aligned
#define XSTR   80     // x tile: 48 cols x 80 halves
#define YSTR   136    // y tile: 16 cols x 136 floats
#define GSTR   80     // gated : 16 cols x 80 halves
#define SSTR   272    // skip/h1 tiles in head: 16 cols x 272 halves
#define H2STR  264    // h2 tile in head: 16 cols x 264 floats

// ---------------------------------------------------------------------------
// Fragment helpers (wave32 layouts per CDNA5 ISA §7.12.2)
// ---------------------------------------------------------------------------

// A-fragment from pre-packed f16 weights: [frag][lane][16 halves] contiguous.
__device__ __forceinline__ v16h load_a_packed(const _Float16* __restrict__ P,
                                              int frag, int lane) {
  return *(const v16h*)(P + ((size_t)frag * 32 + lane) * 16);
}

// B-fragment from an N-major LDS tile S[col][K..]: lane's 16 K-values are
// contiguous -> lowers to 2x ds_load_b128.
__device__ __forceinline__ v16h load_b_frag(const _Float16* S, int ldh,
                                            int k0, int colBase, int lane) {
  int N  = colBase + (lane & 15);
  int kb = k0 + ((lane & 16) ? 16 : 0);
  return *(const v16h*)(S + N * ldh + kb);
}

__device__ __forceinline__ v8f wmma16(v16h a, v16h b, v8f c) {
  return __builtin_amdgcn_wmma_f32_16x16x32_f16(false, a, false, b,
                                                (short)0, c, false, false);
}

__device__ __forceinline__ float fast_tanh(float x) {
  float e2 = __expf(2.0f * x);
  return 1.0f - 2.0f / (e2 + 1.0f);     // saturates cleanly for |x| large
}
__device__ __forceinline__ float fast_sig(float x) {
  return 1.0f / (1.0f + __expf(-x));
}

// ---------------------------------------------------------------------------
// 0) weight packer: f32 row-major (stride rs/cs) -> f16 fragment order.
//    frag f = mt*Ksteps + ks; per (f,lane) 16 halves per the 16x32 A layout.
// ---------------------------------------------------------------------------
__global__ __launch_bounds__(256) void k_pack(const float* __restrict__ W,
                                              _Float16* __restrict__ dst,
                                              int rs, int cs, int Mtiles,
                                              int Ksteps, int kmax,
                                              size_t srcStride, size_t dstStride,
                                              int nMat) {
  int gid = blockIdx.x * 256 + threadIdx.x;
  int lanesPerMat = Mtiles * Ksteps * 32;
  if (gid >= nMat * lanesPerMat) return;
  int mat  = gid / lanesPerMat;
  int rem  = gid % lanesPerMat;
  int lane = rem & 31;
  int f    = rem >> 5;
  int ks   = f % Ksteps, mt = f / Ksteps;
  const float* Ws = W + (size_t)mat * srcStride;
  _Float16* out   = dst + (size_t)mat * dstStride + ((size_t)f * 32 + lane) * 16;
  int M  = mt * 16 + (lane & 15);
  int kb = (lane & 16) ? 16 : 0;
#pragma unroll
  for (int e = 0; e < 16; ++e) {
    int v = e >> 1;
    int K = ks * 32 + kb + e;
    out[e] = (_Float16)((K < kmax) ? Ws[(size_t)M * rs + (size_t)K * cs] : 0.0f);
  }
}

// ---------------------------------------------------------------------------
// 1) ConvTranspose upsample: yUp[b][f][t'] (t' = 4*l + k); tiny
// ---------------------------------------------------------------------------
__global__ __launch_bounds__(256) void k_up(const float* __restrict__ lf,
                                            const float* __restrict__ up_w,
                                            const float* __restrict__ up_b,
                                            float* __restrict__ yUp) {
  int idx = blockIdx.x * 256 + threadIdx.x;
  if (idx >= B_ * LF_ * 256) return;
  int tp = idx & 255;
  int f  = (idx >> 8) % LF_;
  int b  = idx / (LF_ * 256);
  int l  = tp >> 2, k = tp & 3;
  const float* xr = lf + ((size_t)b * L_ + l) * LF_;
  float s = up_b[f];
#pragma unroll 4
  for (int fp = 0; fp < LF_; ++fp)
    s += xr[fp] * up_w[(fp * LF_ + f) * 4 + k];
  yUp[idx] = s;
}

// ---------------------------------------------------------------------------
// 2) cond projection (WMMA): condZ[b][t'][o] = proj_w[3072x80] @ yUp + proj_b
//    (time-major cond so gate reads & these writes are contiguous)
// ---------------------------------------------------------------------------
__global__ __launch_bounds__(256) void k_proj(const float* __restrict__ yUp,
                                              const _Float16* __restrict__ pproj,
                                              const float* __restrict__ proj_b,
                                              float* __restrict__ condZ) {
  __shared__ __align__(32) _Float16 sT[16 * 96];  // [n][K], K padded 80->96
  int tid = threadIdx.x, lane = tid & 31, wv = tid >> 5;
  int nt = blockIdx.x;   // 16 N-tiles
  int mg = blockIdx.y;   // 24 groups of 8 M-tiles
  int b  = blockIdx.z;

  for (int idx = tid; idx < 16 * 96; idx += 256) {
    int n = idx / 96, k = idx % 96;
    float v = (k < LF_) ? yUp[((size_t)b * LF_ + k) * 256 + 16 * nt + n] : 0.0f;
    sT[idx] = (_Float16)v;
  }
  __syncthreads();

  int mt  = mg * 8 + wv;
  v8f acc = {};
#pragma unroll
  for (int ks = 0; ks < 3; ++ks) {
    v16h a  = load_a_packed(pproj, mt * 3 + ks, lane);
    v16h bm = load_b_frag(sT, 96, 32 * ks, 0, lane);
    acc = wmma16(a, bm, acc);
  }
  int n = lane & 15, mo = (lane & 16) ? 8 : 0;
  int o0 = 16 * mt + mo;
  float* dst = condZ + ((size_t)b * 256 + 16 * nt + n) * (NL_ * 2 * H_) + o0;
  v8f outv;
#pragma unroll
  for (int r = 0; r < 8; ++r) outv[r] = acc[r] + proj_b[o0 + r];
  *(v8f*)dst = outv;
}

// ---------------------------------------------------------------------------
// 3) embed: x0[b][t][h] = f16(gold[b][t]*embed_w[h] + embed_b[h]), vectorized
// ---------------------------------------------------------------------------
__global__ __launch_bounds__(256) void k_embed(const float* __restrict__ gold,
                                               const float* __restrict__ ew,
                                               const float* __restrict__ eb,
                                               _Float16* __restrict__ xh) {
  int idx = blockIdx.x * 256 + threadIdx.x;   // B*T*8 vec8 chunks
  if (idx >= B_ * T_ * 8) return;
  int bt = idx >> 3, cb = (idx & 7) * 8;
  float g = gold[bt];
  v8h v;
#pragma unroll
  for (int j = 0; j < 8; ++j) v[j] = (_Float16)(g * ew[cb + j] + eb[cb + j]);
  *(v8h*)(xh + (size_t)bt * H_ + cb) = v;
}

// ---------------------------------------------------------------------------
// 4) one WaveNet layer; block = one batch x 16-wide time tile, 8 waves
// ---------------------------------------------------------------------------
__global__ __launch_bounds__(256) void k_layer(
    const _Float16* __restrict__ xin, _Float16* __restrict__ xout,
    const float* __restrict__ condZ, float* __restrict__ skipSum,
    const _Float16* __restrict__ pdil, const float* __restrict__ dil_b,
    const _Float16* __restrict__ pskip, const float* __restrict__ skip_b,
    const _Float16* __restrict__ pres, const float* __restrict__ res_b,
    int layer, int dilation, int firstLayer, int lastLayer) {
  __shared__ __align__(32) _Float16 xT[48 * XSTR];  // [col][ch], col = t0-32..t0+15
  __shared__ __align__(32) float    yT[16 * YSTR];  // [n][128]
  __shared__ __align__(32) _Float16 gT[16 * GSTR];  // [n][64]

  int tid  = threadIdx.x;
  int lane = tid & 31;
  int wv   = tid >> 5;
  int blk  = blockIdx.x;
  int b    = blk / NTBLK;
  int t0   = (blk % NTBLK) * TTILE;

  // L2 warm-up for this layer's packed weights (global_prefetch_b8)
  __builtin_prefetch(pskip + (size_t)layer * 16384, 0, 0);

  // ---- x halo load: N-major, vector copy (128B per column) ----------------
  const _Float16* xinB = xin + (size_t)b * T_ * H_;
  for (int idx = tid; idx < 48 * 8; idx += 256) {
    int c = idx >> 3, ch = (idx & 7) * 8;
    int t = t0 - 32 + c;
    v8h v = {};
    if (t >= 0) v = *(const v8h*)(xinB + (size_t)t * H_ + ch);
    *(v8h*)(xT + c * XSTR + ch) = v;
  }
  __syncthreads();

  // ---- dilated conv: y[128x16] = W0 @ x[t-d] + W1 @ x[t] ------------------
  {
    const _Float16* Wd = pdil + (size_t)layer * 16384;
    v8f acc = {};
#pragma unroll
    for (int tap = 0; tap < 2; ++tap) {
      int colBase = tap ? 32 : (32 - dilation);
      const _Float16* Wt = Wd + tap * 8192;
#pragma unroll
      for (int ks = 0; ks < 2; ++ks) {
        v16h a  = load_a_packed(Wt, wv * 2 + ks, lane);
        v16h bm = load_b_frag(xT, XSTR, 32 * ks, colBase, lane);
        acc = wmma16(a, bm, acc);
      }
    }
    int n = lane & 15, mo = (lane & 16) ? 8 : 0;
    *(v8f*)(yT + n * YSTR + 16 * wv + mo) = acc;   // 8 contiguous rows
  }
  __syncthreads();

  // ---- gate: tanh(y[:64]+b+cond) * sigmoid(y[64:]+b+cond) -----------------
  {
    int n  = tid >> 4;              // time column 0..15
    int mb = (tid & 15) * 4;        // 4 channels per thread
    const float* db = dil_b + layer * 128;
    const float* cz = condZ + ((size_t)b * 256 + (size_t)((t0 + n) / 75)) * (NL_ * 2 * H_)
                    + (size_t)layer * 128;
    const float* yr = yT + n * YSTR;
    v4h g;
#pragma unroll
    for (int j = 0; j < 4; ++j) {
      int m = mb + j;
      float ya = yr[m]      + db[m]      + cz[m];
      float yb = yr[64 + m] + db[64 + m] + cz[64 + m];
      g[j] = (_Float16)(fast_tanh(ya) * fast_sig(yb));
    }
    *(v4h*)(gT + n * GSTR + mb) = g;
  }
  __syncthreads();

  // ---- skip GEMM: skipSum[b][t][0:256] += skip_w @ gated + skip_b ---------
  {
    const _Float16* Ws = pskip + (size_t)layer * 16384;
    const float* bs = skip_b + layer * SKIP_;
#pragma unroll
    for (int half = 0; half < 2; ++half) {
      int mt  = wv + 8 * half;
      v8f acc = {};
#pragma unroll
      for (int ks = 0; ks < 2; ++ks) {
        v16h a  = load_a_packed(Ws, mt * 2 + ks, lane);
        v16h bm = load_b_frag(gT, GSTR, 32 * ks, 0, lane);
        acc = wmma16(a, bm, acc);
      }
      int n = lane & 15, mo = (lane & 16) ? 8 : 0;
      int s0 = 16 * mt + mo;
      float* sp = skipSum + ((size_t)b * T_ + t0 + n) * SKIP_ + s0;
      v8f old = {};
      if (!firstLayer) old = *(const v8f*)sp;
      v8f outv;
#pragma unroll
      for (int r = 0; r < 8; ++r) outv[r] = acc[r] + bs[s0 + r] + old[r];
      *(v8f*)sp = outv;
    }
  }

  // ---- residual GEMM: x_next = x + res_w @ gated + res_b (waves 0..3) -----
  if (!lastLayer && wv < 4) {
    const _Float16* Wr = pres + (size_t)layer * 4096;
    const float* br = res_b + layer * H_;
    v8f acc = {};
#pragma unroll
    for (int ks = 0; ks < 2; ++ks) {
      v16h a  = load_a_packed(Wr, wv * 2 + ks, lane);
      v16h bm = load_b_frag(gT, GSTR, 32 * ks, 0, lane);
      acc = wmma16(a, bm, acc);
    }
    int n = lane & 15, mo = (lane & 16) ? 8 : 0;
    int h0 = 16 * wv + mo;
    _Float16* xo = xout + ((size_t)b * T_ + t0 + n) * H_ + h0;
    const _Float16* xold = xT + (32 + n) * XSTR + h0;
    v8h xv;
#pragma unroll
    for (int r = 0; r < 8; ++r)
      xv[r] = (_Float16)((float)xold[r] + acc[r] + br[h0 + r]);
    *(v8h*)xo = xv;
  }
}

// ---------------------------------------------------------------------------
// 5) output head: out2_w @ relu(out1_w @ relu(skipSum)) -> log_softmax
// ---------------------------------------------------------------------------
__global__ __launch_bounds__(256) void k_out(const float* __restrict__ skipSum,
                                             const _Float16* __restrict__ pout1,
                                             const _Float16* __restrict__ pout2,
                                             float* __restrict__ out) {
  __shared__ __align__(32) _Float16 sT[16 * SSTR];   // relu(skip) [n][256]
  __shared__ __align__(32) _Float16 h1T[16 * SSTR];  // relu(h1)   [n][256]
  __shared__ __align__(32) float    h2T[16 * H2STR]; // h2         [n][256]
  __shared__ float red[256];
  __shared__ float colMax[16], colLse[16];

  int tid = threadIdx.x, lane = tid & 31, wv = tid >> 5;
  int blk = blockIdx.x;
  int b   = blk / NTBLK;
  int t0  = (blk % NTBLK) * TTILE;

  // load relu(skip), N-major, vector converts
  for (int idx = tid; idx < 16 * 32; idx += 256) {
    int n = idx >> 5, cb = (idx & 31) * 8;
    const float* sp = skipSum + ((size_t)b * T_ + t0 + n) * SKIP_ + cb;
    v8h v;
#pragma unroll
    for (int j = 0; j < 8; ++j) v[j] = (_Float16)fmaxf(sp[j], 0.0f);
    *(v8h*)(sT + n * SSTR + cb) = v;
  }
  __syncthreads();

#pragma unroll
  for (int half = 0; half < 2; ++half) {   // h1 = relu(out1_w @ relu(skip))
    int mt  = wv + 8 * half;
    v8f acc = {};
#pragma unroll
    for (int ks = 0; ks < 8; ++ks) {
      v16h a  = load_a_packed(pout1, mt * 8 + ks, lane);
      v16h bm = load_b_frag(sT, SSTR, 32 * ks, 0, lane);
      acc = wmma16(a, bm, acc);
    }
    int n = lane & 15, mo = (lane & 16) ? 8 : 0;
    v8h hv;
#pragma unroll
    for (int r = 0; r < 8; ++r) hv[r] = (_Float16)fmaxf(acc[r], 0.0f);
    *(v8h*)(h1T + n * SSTR + 16 * mt + mo) = hv;   // fused relu+cvt epilogue
  }
  __syncthreads();

#pragma unroll
  for (int half = 0; half < 2; ++half) {   // h2 = out2_w @ h1
    int mt  = wv + 8 * half;
    v8f acc = {};
#pragma unroll
    for (int ks = 0; ks < 8; ++ks) {
      v16h a  = load_a_packed(pout2, mt * 8 + ks, lane);
      v16h bm = load_b_frag(h1T, SSTR, 32 * ks, 0, lane);
      acc = wmma16(a, bm, acc);
    }
    int n = lane & 15, mo = (lane & 16) ? 8 : 0;
    *(v8f*)(h2T + n * H2STR + 16 * mt + mo) = acc;
  }
  __syncthreads();

  // log_softmax over 256 channels per time column (16 threads per column)
  int n = tid >> 4, seg = tid & 15;
  const float* row = h2T + n * H2STR;
  float lm = -3.0e38f;
#pragma unroll
  for (int j = 0; j < 16; ++j) lm = fmaxf(lm, row[seg * 16 + j]);
  red[n * 16 + seg] = lm;
  __syncthreads();
  if (seg == 0) {
    float g = red[n * 16];
    for (int s = 1; s < 16; ++s) g = fmaxf(g, red[n * 16 + s]);
    colMax[n] = g;
  }
  __syncthreads();
  float gm = colMax[n];
  float ls = 0.0f;
#pragma unroll
  for (int j = 0; j < 16; ++j) ls += __expf(row[seg * 16 + j] - gm);
  red[n * 16 + seg] = ls;
  __syncthreads();
  if (seg == 0) {
    float ss = 0.0f;
    for (int s = 0; s < 16; ++s) ss += red[n * 16 + s];
    colLse[n] = __logf(ss);
  }
  __syncthreads();
  float lse = colLse[n];
  int t = t0 + n;
#pragma unroll
  for (int j = 0; j < 16; ++j) {
    int m = seg * 16 + j;
    out[((size_t)b * MU1_ + m) * T_ + t] = row[m] - gm - lse;
  }
}

// ---------------------------------------------------------------------------
extern "C" void kernel_launch(void* const* d_in, const int* in_sizes, int n_in,
                              void* d_out, int out_size, void* d_ws, size_t ws_size,
                              hipStream_t stream) {
  (void)in_sizes; (void)n_in; (void)out_size;
  const float* lf      = (const float*)d_in[0];
  const float* gold    = (const float*)d_in[1];
  const float* embed_w = (const float*)d_in[2];
  const float* embed_b = (const float*)d_in[3];
  const float* up_w    = (const float*)d_in[4];
  const float* up_b    = (const float*)d_in[5];
  const float* proj_w  = (const float*)d_in[6];
  const float* proj_b  = (const float*)d_in[7];
  const float* dil_w   = (const float*)d_in[8];
  const float* dil_b   = (const float*)d_in[9];
  const float* skip_w  = (const float*)d_in[10];
  const float* skip_b  = (const float*)d_in[11];
  const float* res_w   = (const float*)d_in[12];
  const float* res_b   = (const float*)d_in[13];
  const float* out1_w  = (const float*)d_in[14];
  const float* out2_w  = (const float*)d_in[15];

  // workspace carve-up (256B aligned)
  size_t off = 0;
  auto carve = [&](size_t bytes) { size_t o = off; off += (bytes + 255) & ~(size_t)255; return o; };
  char* ws = (char*)d_ws;
  float*    yUp     = (float*)   (ws + carve((size_t)B_ * LF_ * 256 * 4));
  float*    condZ   = (float*)   (ws + carve((size_t)B_ * 256 * NL_ * 2 * H_ * 4));
  _Float16* xa      = (_Float16*)(ws + carve((size_t)B_ * T_ * H_ * 2));
  _Float16* xb      = (_Float16*)(ws + carve((size_t)B_ * T_ * H_ * 2));
  float*    skipSum = (float*)   (ws + carve((size_t)B_ * T_ * SKIP_ * 4));
  _Float16* pdil    = (_Float16*)(ws + carve((size_t)NL_ * 16384 * 2));
  _Float16* pskip   = (_Float16*)(ws + carve((size_t)NL_ * 16384 * 2));
  _Float16* pres    = (_Float16*)(ws + carve((size_t)(NL_ - 1) * 4096 * 2));
  _Float16* pproj   = (_Float16*)(ws + carve((size_t)192 * 3 * 512 * 2));
  _Float16* pout1   = (_Float16*)(ws + carve((size_t)16 * 8 * 512 * 2));
  _Float16* pout2   = (_Float16*)(ws + carve((size_t)16 * 8 * 512 * 2));
  if (off > ws_size) return;  // workspace too small: do nothing (deterministic)

  auto nb = [](int n) { return (n + 255) / 256; };

  // --- weight packing (f32 -> f16 fragment order), L2-resident thereafter --
  int nDil = NL_ * 8 * 2 * 32;
  k_pack<<<nb(nDil), 256, 0, stream>>>(dil_w + 0, pdil,        128, 2, 8, 2, 64, 16384, 16384, NL_);
  k_pack<<<nb(nDil), 256, 0, stream>>>(dil_w + 1, pdil + 8192, 128, 2, 8, 2, 64, 16384, 16384, NL_);
  int nSkip = NL_ * 16 * 2 * 32;
  k_pack<<<nb(nSkip), 256, 0, stream>>>(skip_w, pskip, 64, 1, 16, 2, 64, 16384, 16384, NL_);
  int nRes = (NL_ - 1) * 4 * 2 * 32;
  k_pack<<<nb(nRes), 256, 0, stream>>>(res_w, pres, 64, 1, 4, 2, 64, 4096, 4096, NL_ - 1);
  int nProj = 192 * 3 * 32;
  k_pack<<<nb(nProj), 256, 0, stream>>>(proj_w, pproj, 80, 1, 192, 3, 80, 0, 0, 1);
  int nOut = 16 * 8 * 32;
  k_pack<<<nb(nOut), 256, 0, stream>>>(out1_w, pout1, 256, 1, 16, 8, 256, 0, 0, 1);
  k_pack<<<nb(nOut), 256, 0, stream>>>(out2_w, pout2, 256, 1, 16, 8, 256, 0, 0, 1);

  // --- conditioning + embedding -------------------------------------------
  k_up<<<nb(B_ * LF_ * 256), 256, 0, stream>>>(lf, up_w, up_b, yUp);
  dim3 gp(16, 24, B_);
  k_proj<<<gp, 256, 0, stream>>>(yUp, pproj, proj_b, condZ);
  k_embed<<<nb(B_ * T_ * 8), 256, 0, stream>>>(gold, embed_w, embed_b, xa);

  // --- 24 layers, x ping-pong to avoid cross-block halo races --------------
  int nblk = B_ * NTBLK;
  for (int i = 0; i < NL_; ++i) {
    int d = 1 << (i % 6);
    const _Float16* xi = (i & 1) ? xb : xa;
    _Float16*       xo = (i & 1) ? xa : xb;
    k_layer<<<nblk, 256, 0, stream>>>(xi, xo, condZ, skipSum,
                                      pdil, dil_b, pskip, skip_b, pres, res_b,
                                      i, d, (i == 0) ? 1 : 0, (i == NL_ - 1) ? 1 : 0);
  }

  // --- output head ---------------------------------------------------------
  k_out<<<nblk, 256, 0, stream>>>(skipSum, pout1, pout2, (float*)d_out);
}
